// PerformerAttentionReLU_79465484911160
// MI455X (gfx1250) — matrix operand order repaired and checked by reading, not compiled, not run on hardware
//
#include <hip/hip_runtime.h>
#include <hip/hip_bf16.h>

// ---------------------------------------------------------------------------
// Performer (ReLU random-feature linear attention) for MI455X / gfx1250.
// All contractions on v_wmma_f32_16x16x32_bf16 (fp32 accumulate).
// Data movement: global_load_async_to_lds_b128 (ASYNCcnt) + LDS double buffer.
// Layout rule: every operand is stored with the contraction index innermost
// ([rows][K]); transposed outputs use the packed-b128 epilogue (lane = column,
// regs = 8 consecutive rows -> one 16B store).
// ---------------------------------------------------------------------------

typedef __bf16 bf16;
typedef __attribute__((ext_vector_type(16))) __bf16 v16bf;
typedef __attribute__((ext_vector_type(8)))  __bf16 v8bf;
typedef __attribute__((ext_vector_type(8)))  float  v8f;

constexpr int Bb = 16, Nn = 4096, Cc = 768, Hh = 12, Dd = 64, Mm = 256;
constexpr int Rows = Bb * Nn;      // 65536
constexpr int QKVC = 3 * Cc;       // 2304
constexpr int QKC  = 2 * Cc;       // 1536 (q|k block width)

constexpr int BM = 128, BN = 64, BK = 32;
constexpr int PAD = 8;             // rows are 80B -> every 8-elem chunk stays 16B aligned

enum { MODE_QKV = 0, MODE_PHIQ, MODE_PHIKT, MODE_KV, MODE_ATTN, MODE_PROJ };

// --- WMMA fragment loaders (CDNA5 ISA 7.12.2, wave32) -----------------------
// A (16x32 bf16): lanes 0-15 -> row=lane, K = 0..15 ; lanes 16-31 -> K = 16..31
__device__ __forceinline__ v16bf load_frag_a(const bf16* src, int ld, int lane) {
  const bf16* p = src + (size_t)(lane & 15) * ld + ((lane >> 4) ? 16 : 0);
  v16bf a;
#pragma unroll
  for (int i = 0; i < 16; ++i) a[i] = p[i];
  return a;
}
// B (32x16 bf16) read from BT[N][K]: lane = column N, K contiguous (halves 0..15 / 16..31)
__device__ __forceinline__ v16bf load_frag_bt(const bf16* src, int ld, int lane) {
  const bf16* p = src + (size_t)(lane & 15) * ld + ((lane >> 4) ? 16 : 0);
  v16bf b;
#pragma unroll
  for (int i = 0; i < 16; ++i) b[i] = p[i];
  return b;
}

// Async 16B global -> LDS copy (VGLOBAL GLOBAL_LOAD_ASYNC_TO_LDS_B128, GV mode).
// INST_OFFSET applies to both the global and the LDS address (ISA 10.7 / 15.18.3).
__device__ __forceinline__ void async_copy_b128(const bf16* gaddr, unsigned lds_addr) {
  asm volatile("global_load_async_to_lds_b128 %0, %1, off"
               :: "v"(lds_addr), "v"(gaddr) : "memory");
}
__device__ __forceinline__ void async_copy_b128_off16(const bf16* gaddr, unsigned lds_addr) {
  asm volatile("global_load_async_to_lds_b128 %0, %1, off offset:16"
               :: "v"(lds_addr), "v"(gaddr) : "memory");
}
__device__ __forceinline__ void wait_async0() {
  asm volatile("s_wait_asynccnt 0x0" ::: "memory");
}

// ---------------------------------------------------------------------------
template <int MODE>
__global__ __launch_bounds__(256)
void gemm_wmma(const bf16* __restrict__ Aall, const bf16* __restrict__ Ball,
               void* __restrict__ Cout, void* __restrict__ Cout2,
               const float* __restrict__ bias)
{
  __shared__ bf16 sA[2][BM][BK + PAD];   // [m][k]
  __shared__ bf16 sB[2][BN][BK + PAD];   // BT: [n][k]

  const int t    = threadIdx.x;
  const int lane = t & 31;
  const int wave = t >> 5;
  const int wm   = wave >> 1;            // 0..3
  const int wn   = wave & 1;             // 0..1

  const int z = blockIdx.z;
  const int bh_b = z / Hh, bh_h = z % Hh;
  (void)bh_b; (void)bh_h;

  const bf16* A; const bf16* BT; int lda, ldb, K;
  if constexpr (MODE == MODE_QKV) {              // [65536,768] @ W_qkv -> qk | vT
    lda = Cc; ldb = Cc; K = Cc;
    A  = Aall + (size_t)blockIdx.y * BM * lda;
    BT = Ball;                                   // WqkvT [2304][768]
  } else if constexpr (MODE == MODE_PHIQ || MODE == MODE_PHIKT) {
    lda = QKC; ldb = Dd; K = Dd;                 // per (b,h): [4096,64] @ rf^T
    const int coff = (MODE == MODE_PHIQ ? 0 : Cc) + bh_h * Dd;
    A  = Aall + ((size_t)bh_b * Nn + (size_t)blockIdx.y * BM) * QKC + coff;
    BT = Ball;                                   // rf [256][64] (already K-innermost)
  } else if constexpr (MODE == MODE_KV) {
    lda = Nn; ldb = Nn; K = Nn;                  // per (b,h): [256,4096] @ v
    A  = Aall + (size_t)z * Mm * Nn + (size_t)blockIdx.y * BM * Nn;   // phi_k^T [m][n]
    BT = Ball + (size_t)z * Dd * Nn;                                  // vT [d][n]
  } else if constexpr (MODE == MODE_ATTN) {
    lda = Mm; ldb = Mm; K = Mm;                  // per (b,h): [4096,256] @ kv
    A  = Aall + (size_t)z * Nn * Mm + (size_t)blockIdx.y * BM * Mm;   // phi_q [n][m]
    BT = Ball + (size_t)z * Dd * Mm;                                  // kvT [d][m]
  } else {                                       // PROJ: [65536,768] @ W_proj
    lda = Cc; ldb = Cc; K = Cc;
    A  = Aall + (size_t)blockIdx.y * BM * lda;
    BT = Ball;                                   // WprojT [768][768]
  }
  const int nblockbase = blockIdx.x * BN;

  v8f acc[2][2] = {};

  // Cooperative async staging: A tile 128x32 (16 elem/thread), BT tile 64x32 (8/thread).
  const int arow = t >> 1, acol = (t & 1) * 16;
  const int brow = t >> 2, bcol = (t & 3) * 8;

  auto issue_tiles = [&](int k0, int buf) {
    const bf16* ga = A + (size_t)arow * lda + k0 + acol;
    unsigned la = (unsigned)(size_t)&sA[buf][arow][acol];
    async_copy_b128(ga, la);
    async_copy_b128_off16(ga, la);               // second 8-elem chunk (offset both sides)
    const bf16* gb = BT + (size_t)(nblockbase + brow) * ldb + k0 + bcol;
    unsigned lb = (unsigned)(size_t)&sB[buf][brow][bcol];
    async_copy_b128(gb, lb);
  };

  issue_tiles(0, 0);
  int cur = 0;
  for (int k0 = 0; k0 < K; k0 += BK) {
    wait_async0();          // own copies into sX[cur] complete
    __syncthreads();        // everyone's copies visible; prior reads of sX[cur^1] done
    if (k0 + BK < K) issue_tiles(k0 + BK, cur ^ 1);

    v16bf fa[2], fb[2];
#pragma unroll
    for (int i = 0; i < 2; ++i) fa[i] = load_frag_a(&sA[cur][wm * 32 + i * 16][0], BK + PAD, lane);
#pragma unroll
    for (int j = 0; j < 2; ++j) fb[j] = load_frag_bt(&sB[cur][wn * 32 + j * 16][0], BK + PAD, lane);
#pragma unroll
    for (int i = 0; i < 2; ++i)
#pragma unroll
      for (int j = 0; j < 2; ++j)
        acc[i][j] = __builtin_amdgcn_wmma_f32_16x16x32_bf16(
            false, fa[i], false, fb[j], (short)0, acc[i][j], false, false);
    cur ^= 1;
  }

  // --- Epilogues. C layout: lane 0-15 -> col=lane, reg r -> row=r (+8 lanes 16-31).
  const float qscale = 0.125f;          // D^-1/2
  const float mscale = 1.0f / (float)Mm;
  const int rbase = (lane >> 4) ? 8 : 0;

#pragma unroll
  for (int i = 0; i < 2; ++i) {
#pragma unroll
    for (int j = 0; j < 2; ++j) {
      const int mrow  = blockIdx.y * BM + wm * 32 + i * 16 + rbase;
      const int jbase = nblockbase + wn * 32 + j * 16;
      const int ncol  = jbase + (lane & 15);
      v8f c = acc[i][j];
      if constexpr (MODE == MODE_QKV) {
        const float bv = bias[ncol];
        if (jbase < QKC) {              // q|k -> row-major qkb [Rows][1536]
          bf16* C = (bf16*)Cout;
#pragma unroll
          for (int r = 0; r < 8; ++r)
            C[(size_t)(mrow + r) * QKC + ncol] = (bf16)(c[r] + bv);
        } else {                        // v -> transposed vT [b*H+h][d][n] (packed b128)
          bf16* V = (bf16*)Cout2;
          const int jc = ncol - QKC, h = jc >> 6, d = jc & 63;
          const int bb = mrow >> 12, n = mrow & (Nn - 1);
          v8bf o;
#pragma unroll
          for (int r = 0; r < 8; ++r) o[r] = (bf16)(c[r] + bv);
          *reinterpret_cast<v8bf*>(&V[((size_t)(bb * Hh + h) * Dd + d) * Nn + n]) = o;
        }
      } else if constexpr (MODE == MODE_PHIQ) {   // [n][m] row-major
        bf16* C = (bf16*)Cout + (size_t)z * Nn * Mm;
#pragma unroll
        for (int r = 0; r < 8; ++r)
          C[(size_t)(mrow + r) * Mm + ncol] = (bf16)fmaxf(c[r] * qscale, 0.0f);
      } else if constexpr (MODE == MODE_PHIKT) {  // [m][n] transposed, packed b128
        bf16* C = (bf16*)Cout + (size_t)z * Mm * Nn;
        v8bf o;
#pragma unroll
        for (int r = 0; r < 8; ++r) o[r] = (bf16)fmaxf(c[r] * qscale, 0.0f);
        *reinterpret_cast<v8bf*>(&C[(size_t)ncol * Nn + mrow]) = o;
      } else if constexpr (MODE == MODE_KV) {     // kvT [d][m] transposed, packed b128
        bf16* C = (bf16*)Cout + (size_t)z * Dd * Mm;
        v8bf o;
#pragma unroll
        for (int r = 0; r < 8; ++r) o[r] = (bf16)c[r];
        *reinterpret_cast<v8bf*>(&C[(size_t)ncol * Mm + mrow]) = o;
      } else if constexpr (MODE == MODE_ATTN) {   // attnb [b*N+n][H*D] row-major
        bf16* C = (bf16*)Cout + (size_t)bh_b * Nn * Cc + bh_h * Dd;
#pragma unroll
        for (int r = 0; r < 8; ++r)
          C[(size_t)(mrow + r) * Cc + ncol] = (bf16)(c[r] * mscale);
      } else {                                    // PROJ -> fp32 out + bias
        float* C = (float*)Cout;
        const float bv = bias[ncol];
#pragma unroll
        for (int r = 0; r < 8; ++r)
          C[(size_t)(mrow + r) * Cc + ncol] = c[r] + bv;
      }
    }
  }
}

// --- conversion kernels -----------------------------------------------------
__global__ __launch_bounds__(256)
void cvt_f32_bf16(const float* __restrict__ in, bf16* __restrict__ out, int n) {
  int i = blockIdx.x * blockDim.x + threadIdx.x;
  if (i < n) out[i] = (bf16)in[i];
}
__global__ __launch_bounds__(256)
void cvt_transpose_bf16(const float* __restrict__ in, bf16* __restrict__ out, int R, int C) {
  int i = blockIdx.x * blockDim.x + threadIdx.x;   // in [R][C] -> out [C][R]
  if (i < R * C) {
    int r = i / C, c = i % C;
    out[(size_t)c * R + r] = (bf16)in[i];
  }
}

// ---------------------------------------------------------------------------
extern "C" void kernel_launch(void* const* d_in, const int* in_sizes, int n_in,
                              void* d_out, int out_size, void* d_ws, size_t ws_size,
                              hipStream_t stream)
{
  (void)in_sizes; (void)n_in; (void)out_size; (void)ws_size;
  const float* x     = (const float*)d_in[0];   // [16,4096,768]
  const float* Wqkv  = (const float*)d_in[1];   // [768,2304]
  const float* bqkv  = (const float*)d_in[2];   // [2304]
  const float* Wproj = (const float*)d_in[3];   // [768,768]
  const float* bproj = (const float*)d_in[4];   // [768]
  const float* rf    = (const float*)d_in[5];   // [256,64]  (already D-innermost)

  char* ws = (char*)d_ws;
  size_t off = 0;
  auto carve = [&](size_t bytes) -> void* {
    void* p = ws + off;
    off += (bytes + 255) & ~(size_t)255;
    return p;
  };
  bf16* xb     = (bf16*)carve((size_t)Rows * Cc * 2);               // 100 MB
  bf16* WqkvT  = (bf16*)carve((size_t)QKVC * Cc * 2);               // 3.5 MB  [2304][768]
  bf16* WprojT = (bf16*)carve((size_t)Cc * Cc * 2);                 // 1.2 MB  [768][768]
  bf16* rfb    = (bf16*)carve((size_t)Mm * Dd * 2);                 // 32 KB   [256][64]
  bf16* qkb    = (bf16*)carve((size_t)Rows * QKC * 2);              // 201 MB  [65536][1536]
  bf16* vT     = (bf16*)carve((size_t)Bb * Hh * Dd * Nn * 2);       // 100 MB  [192][64][4096]
  bf16* phiq   = (bf16*)carve((size_t)Bb * Hh * Nn * Mm * 2);       // 403 MB  [192][4096][256]
  bf16* phikT  = (bf16*)carve((size_t)Bb * Hh * Mm * Nn * 2);       // 403 MB  [192][256][4096]
  bf16* kvT    = (bf16*)carve((size_t)Bb * Hh * Dd * Mm * 2);       // 6.3 MB  [192][64][256]
  bf16* attnb  = (bf16*)carve((size_t)Rows * Cc * 2);               // 100 MB  [65536][768]

  const int T = 256;
  cvt_f32_bf16<<<(Rows * Cc + T - 1) / T, T, 0, stream>>>(x, xb, Rows * Cc);
  cvt_transpose_bf16<<<(Cc * QKVC + T - 1) / T, T, 0, stream>>>(Wqkv, WqkvT, Cc, QKVC);
  cvt_transpose_bf16<<<(Cc * Cc + T - 1) / T, T, 0, stream>>>(Wproj, WprojT, Cc, Cc);
  cvt_f32_bf16<<<(Mm * Dd + T - 1) / T, T, 0, stream>>>(rf, rfb, Mm * Dd);

  // qkv projection: q|k row-major + v transposed, bias fused.
  gemm_wmma<MODE_QKV><<<dim3(QKVC / BN, Rows / BM, 1), T, 0, stream>>>(xb, WqkvT, qkb, vT, bqkv);
  // phi_q = relu(scale * q @ rf^T) stored [n][m]
  gemm_wmma<MODE_PHIQ><<<dim3(Mm / BN, Nn / BM, Bb * Hh), T, 0, stream>>>(qkb, rfb, phiq, nullptr, nullptr);
  // phi_k stored transposed [m][n]
  gemm_wmma<MODE_PHIKT><<<dim3(Mm / BN, Nn / BM, Bb * Hh), T, 0, stream>>>(qkb, rfb, phikT, nullptr, nullptr);
  // kv = phi_k^T @ v, stored transposed kvT [d][m]
  gemm_wmma<MODE_KV><<<dim3(Dd / BN, Mm / BM, Bb * Hh), T, 0, stream>>>(phikT, vT, kvT, nullptr, nullptr);
  // attn = (phi_q @ kv)/M -> [b*N+n][H*D]
  gemm_wmma<MODE_ATTN><<<dim3(Dd / BN, Nn / BM, Bb * Hh), T, 0, stream>>>(phiq, kvT, attnb, nullptr, nullptr);
  // out = attn @ W_proj + b_proj (fp32)
  gemm_wmma<MODE_PROJ><<<dim3(Cc / BN, Rows / BM, 1), T, 0, stream>>>(attnb, WprojT, d_out, nullptr, bproj);
}
